// MOE_82626580841193
// MI455X (gfx1250) — compile-verified, matrix operand order checked by reading, not run on hardware
//
#include <hip/hip_runtime.h>

// ---------------- problem constants ----------------
#define NROWS 4096
#define DIN   1024
#define DOUT  1024
#define NDIS  4
#define NSH   2
#define NPRV  2
#define NTR   16
#define TOPK  2

// meta int layout
#define CNT_T  0    // 16
#define CNT_P  16   // 4
#define OFF_T  32   // 17
#define OFF_P  56   // 5
#define FILL_T 64   // 16
#define FILL_P 80   // 4
#define META_N 96

#define LIST_T_CAP (NROWS*TOPK + NTR*64)   // 9216
#define LIST_P_CAP (NROWS + NDIS*64)       // 4352

typedef __attribute__((ext_vector_type(16))) __bf16 v16bf;
typedef __attribute__((ext_vector_type(8)))  float  v8f;

// hardware RNE float->bf16 (v_cvt_*bf16_f32 on gfx1250)
static __device__ __forceinline__ unsigned short f2bf(float f) {
  union { __bf16 h; unsigned short u; } v;
  v.h = (__bf16)f;
  return v.u;
}
// packed pair convert -> one dword (maps to v_cvt_pk_bf16_f32)
static __device__ __forceinline__ unsigned pack2bf(float lo, float hi) {
  union { __bf16 h[2]; unsigned u; } v;
  v.h[0] = (__bf16)lo; v.h[1] = (__bf16)hi;
  return v.u;
}

// ---------------- routing kernels ----------------
__global__ void init_routing(int* listT, int* listP, int* meta) {
  int i = blockIdx.x * 256 + threadIdx.x;
  if (i < LIST_T_CAP) listT[i] = -1;
  if (i < LIST_P_CAP) listP[i] = -1;
  if (i < META_N)     meta[i]  = 0;
}

__global__ void gate_logits(const float* __restrict__ x, const float* __restrict__ Wg,
                            const float* __restrict__ bg, float* __restrict__ logits) {
  int t = blockIdx.x * blockDim.x + threadIdx.x;      // [0, 4096*16)
  int n = t >> 4, e = t & 15;
  const float* xr = x + (long)n * DIN;
  float s = bg[e];
  for (int k = 0; k < DIN; ++k) s += xr[k] * Wg[k * NTR + e];
  logits[t] = s;
}

__global__ void route_kernel(const float* __restrict__ logits, const int* __restrict__ disease,
                             float* __restrict__ wtop, int* __restrict__ idxtop, int* meta) {
  int n = blockIdx.x * blockDim.x + threadIdx.x;
  if (n >= NROWS) return;
  float l[NTR], mx = -1e30f;
  for (int e = 0; e < NTR; ++e) { l[e] = logits[n * NTR + e]; mx = l[e] > mx ? l[e] : mx; }
  float p[NTR], sum = 0.f;
  for (int e = 0; e < NTR; ++e) { p[e] = __expf(l[e] - mx); sum += p[e]; }
  float inv = 1.f / sum;
  int i0 = 0;
  for (int e = 1; e < NTR; ++e) if (p[e] > p[i0]) i0 = e;
  int i1 = (i0 == 0) ? 1 : 0;
  for (int e = 0; e < NTR; ++e) if (e != i0 && p[e] > p[i1]) i1 = e;
  wtop[n * 2 + 0] = p[i0] * inv;  wtop[n * 2 + 1] = p[i1] * inv;
  idxtop[n * 2 + 0] = i0;         idxtop[n * 2 + 1] = i1;
  atomicAdd(&meta[CNT_T + i0], 1);
  atomicAdd(&meta[CNT_T + i1], 1);
  atomicAdd(&meta[CNT_P + disease[n]], 1);
}

__global__ void offsets_kernel(int* meta) {
  if (threadIdx.x == 0 && blockIdx.x == 0) {
    meta[OFF_T] = 0;
    for (int e = 0; e < NTR; ++e)
      meta[OFF_T + e + 1] = meta[OFF_T + e] + ((meta[CNT_T + e] + 63) & ~63);
    meta[OFF_P] = 0;
    for (int e = 0; e < NDIS; ++e)
      meta[OFF_P + e + 1] = meta[OFF_P + e] + ((meta[CNT_P + e] + 63) & ~63);
  }
}

__global__ void scatter_kernel(const float* __restrict__ wtop, const int* __restrict__ idxtop,
                               const int* __restrict__ disease, int* listT, float* wlistT,
                               int* listP, int* meta) {
  int n = blockIdx.x * blockDim.x + threadIdx.x;
  if (n >= NROWS) return;
  for (int t = 0; t < TOPK; ++t) {
    int e = idxtop[n * 2 + t];
    int pos = atomicAdd(&meta[FILL_T + e], 1);
    int dst = meta[OFF_T + e] + pos;
    listT[dst]  = n * 2 + t;
    wlistT[dst] = wtop[n * 2 + t];
  }
  int d = disease[n];
  int pos = atomicAdd(&meta[FILL_P + d], 1);
  listP[meta[OFF_P + d] + pos] = n;
}

__global__ void combine_kernel(const float* __restrict__ ht, unsigned short* __restrict__ F) {
  long i = (long)blockIdx.x * 256 + threadIdx.x;   // [0, 4096*1024)
  long n = i >> 10; int c = (int)(i & 1023);
  float v = ht[(n * 2) * (long)DOUT + c] + ht[(n * 2 + 1) * (long)DOUT + c];
  F[n * 3072 + 2048 + c] = f2bf(v);
}

// ---------------- WMMA GEMM ----------------
// BRANCH: 0=shared (dense rows, NB=2 relu-sum, out bf16)
//         1=private (gather by disease, NB=2 relu-sum, out bf16)
//         2=transfer (gather by expert, relu, per-row scale, out fp32 per-slot)
//         3=heads (dense, K=3072, A bf16 via async-LDS copy, bias=sum of 3, out fp32)
template <int BRANCH>
__global__ __launch_bounds__(256) void gemm64(
    const void* __restrict__ Av, const float* __restrict__ Bmats,
    const float* __restrict__ bias, void* __restrict__ Outv,
    const int* __restrict__ list, const float* __restrict__ wlist,
    const int* __restrict__ offs, int ldout)
{
  constexpr bool GATHER   = (BRANCH == 1 || BRANCH == 2);
  constexpr bool A_BF16   = (BRANCH == 3);
  constexpr bool RELU     = (BRANCH != 3);
  constexpr bool OUT_BF16 = (BRANCH == 0 || BRANCH == 1);
  constexpr bool SCALE    = (BRANCH == 2);
  constexpr int  NB       = (BRANCH == 0 || BRANCH == 1) ? 2 : 1;
  constexpr int  KTOT     = (BRANCH == 3) ? 3072 : 1024;
  constexpr int  NEXP     = (BRANCH == 1) ? NDIS : NTR;
  constexpr int  BSUM     = (BRANCH == 3) ? 3 : 1;
  constexpr int  LDA      = (BRANCH == 3) ? 3072 : 1024;
  constexpr int  LDS_STRIDE = 40;  // bf16 elems, 80B rows => 16B aligned chunks

  const int tid = threadIdx.x;
  const int c0  = blockIdx.y * 64;
  const int rowStart = blockIdx.x * 64;

  int ex = 0;
  if (GATHER) {
    if (rowStart >= offs[NEXP]) return;
    while (rowStart >= offs[ex + 1]) ++ex;
  }

  __shared__ __align__(16) unsigned short As[64 * LDS_STRIDE];
  __shared__ __align__(16) unsigned short Bs[64 * LDS_STRIDE];
  __shared__ int   rowEnt[64];
  __shared__ float rowW[64];

  if (tid < 64) {
    int e = GATHER ? list[rowStart + tid] : (rowStart + tid);
    rowEnt[tid] = e;
    float wv = 1.0f;
    if (SCALE) wv = (e >= 0) ? wlist[rowStart + tid] : 0.0f;
    rowW[tid] = wv;
  }
  __syncthreads();

  const int lane = tid & 31;
  const int wv   = tid >> 5;     // wave 0..7
  const int rg   = wv & 3;       // 16-row group
  const int cg   = wv >> 2;      // 32-col group
  const int m16  = lane & 15;
  const int kh   = lane >> 4;    // K-half per ISA 16-bit A-fragment layout
  const int mh   = kh * 8;

  // staging thread mapping
  const int lr   = tid >> 2, lcs  = tid & 3;    // A: row 0..63, 8-col segment 0..3
  const int bkp  = tid >> 4, bcs4 = tid & 15;   // B: k-pair 0..15, 4-col segment 0..15

  int aent = rowEnt[lr];
  long arow = (BRANCH == 2) ? ((aent >= 0) ? (long)(aent >> 1) : -1L) : (long)aent;

  const float*          Afp = (const float*)Av;
  const unsigned short* Abf = (const unsigned short*)Av;

  long bexoff = 0;
  if (BRANCH == 1) bexoff = (long)ex * 2 * 1024 * 1024;
  if (BRANCH == 2) bexoff = (long)ex * 1024 * 1024;

  // per-thread fixed LDS staging addresses
  unsigned short* AsDst = &As[lr * LDS_STRIDE + lcs * 8];
  const unsigned asLdsByte = (unsigned)(uintptr_t)AsDst;  // low 32 bits = LDS address

  v8f res0, res1;
  for (int i = 0; i < 8; ++i) { res0[i] = 0.f; res1[i] = 0.f; }

  for (int s = 0; s < NB; ++s) {
    const float* Bm = Bmats + bexoff + (long)s * 1024 * 1024;
    v8f acc0, acc1;
    for (int i = 0; i < 8; ++i) { acc0[i] = 0.f; acc1[i] = 0.f; }

    for (int k0 = 0; k0 < KTOT; k0 += 32) {
      __syncthreads();
      // ---- stage A tile 64x32 -> bf16 LDS ----
      if (A_BF16) {
        // A already bf16: raw 16B async copy global -> LDS (ASYNCcnt path)
        const unsigned short* gp = Abf + arow * LDA + k0 + lcs * 8;
        unsigned long long ga = (unsigned long long)(uintptr_t)gp;
        asm volatile("global_load_async_to_lds_b128 %0, %1, off"
                     :: "v"(asLdsByte), "v"(ga) : "memory");
      } else {
        union { unsigned u32[4]; uint4 q; } tmp;
        if (arow >= 0) {
          float4 f0 = *(const float4*)(Afp + arow * LDA + k0 + lcs * 8);
          float4 f1 = *(const float4*)(Afp + arow * LDA + k0 + lcs * 8 + 4);
          tmp.u32[0] = pack2bf(f0.x, f0.y);
          tmp.u32[1] = pack2bf(f0.z, f0.w);
          tmp.u32[2] = pack2bf(f1.x, f1.y);
          tmp.u32[3] = pack2bf(f1.z, f1.w);
          if (k0 + 32 < KTOT)
            __builtin_prefetch(Afp + arow * LDA + k0 + 32 + lcs * 8, 0, 1);
        } else {
          tmp.q = uint4{0u, 0u, 0u, 0u};
        }
        *(uint4*)AsDst = tmp.q;
      }
      // ---- stage B: k-pairs, transposed -> Bs[col][k], dword stores ----
      {
        const float* br0 = Bm + (long)(k0 + 2 * bkp) * 1024 + c0 + bcs4 * 4;
        float4 f0 = *(const float4*)(br0);
        float4 f1 = *(const float4*)(br0 + 1024);
        float a0[4] = {f0.x, f0.y, f0.z, f0.w};
        float a1[4] = {f1.x, f1.y, f1.z, f1.w};
        for (int j = 0; j < 4; ++j)
          *(unsigned*)(&Bs[(bcs4 * 4 + j) * LDS_STRIDE + 2 * bkp]) = pack2bf(a0[j], a1[j]);
        if (k0 + 32 < KTOT)
          __builtin_prefetch(br0 + 32, 0, 1);
      }
      if (A_BF16)
        asm volatile("s_wait_asynccnt 0x0" ::: "memory");
      __syncthreads();

      // ---- fragments (ISA 16-bit 16x32 layout) ----
      union { v16bf v; uint4 q[2]; } fa, fb0, fb1;
      {
        const unsigned short* ap = &As[(rg * 16 + m16) * LDS_STRIDE + kh * 8];
        fa.q[0] = *(const uint4*)(ap);
        fa.q[1] = *(const uint4*)(ap + 16);
      }
      {
        const unsigned short* bp0 = &Bs[(cg * 32 + m16) * LDS_STRIDE + kh * 8];
        fb0.q[0] = *(const uint4*)(bp0);
        fb0.q[1] = *(const uint4*)(bp0 + 16);
        const unsigned short* bp1 = &Bs[(cg * 32 + 16 + m16) * LDS_STRIDE + kh * 8];
        fb1.q[0] = *(const uint4*)(bp1);
        fb1.q[1] = *(const uint4*)(bp1 + 16);
      }
      acc0 = __builtin_amdgcn_wmma_f32_16x16x32_bf16(false, fa.v, false, fb0.v, (short)0, acc0, false, false);
      acc1 = __builtin_amdgcn_wmma_f32_16x16x32_bf16(false, fa.v, false, fb1.v, (short)0, acc1, false, false);
    }

    // ---- epilogue for this B matrix: bias (+relu) accumulate ----
    long bidx = 0;
    if (BRANCH == 0) bidx = (long)s * 1024;
    if (BRANCH == 1) bidx = (long)(ex * 2 + s) * 1024;
    if (BRANCH == 2) bidx = (long)ex * 1024;
    const float* bb = bias + bidx;
    float b0 = 0.f, b1 = 0.f;
    for (int i = 0; i < BSUM; ++i) {
      b0 += bb[i * 1024 + c0 + cg * 32 + m16];
      b1 += bb[i * 1024 + c0 + cg * 32 + 16 + m16];
    }
    for (int r = 0; r < 8; ++r) {
      float v0 = acc0[r] + b0;
      float v1 = acc1[r] + b1;
      if (RELU) { v0 = v0 > 0.f ? v0 : 0.f; v1 = v1 > 0.f ? v1 : 0.f; }
      res0[r] += v0; res1[r] += v1;
    }
  }

  if (SCALE) {
    for (int r = 0; r < 8; ++r) {
      float s = rowW[rg * 16 + r + mh];
      res0[r] *= s; res1[r] *= s;
    }
  }

  // ---- store (C/D layout: VGPR r -> M=r / r+8; lane&15 -> N) ----
  for (int r = 0; r < 8; ++r) {
    int M = rg * 16 + r + mh;
    int e = rowEnt[M];
    if (GATHER && e < 0) continue;
    long R  = e;
    int  cA = c0 + cg * 32 + m16;
    int  cB = cA + 16;
    if (OUT_BF16) {
      unsigned short* O = (unsigned short*)Outv;
      O[R * (long)ldout + cA] = f2bf(res0[r]);
      O[R * (long)ldout + cB] = f2bf(res1[r]);
    } else {
      float* O = (float*)Outv;
      O[R * (long)ldout + cA] = res0[r];
      O[R * (long)ldout + cB] = res1[r];
    }
  }
}

// ---------------- launcher ----------------
extern "C" void kernel_launch(void* const* d_in, const int* in_sizes, int n_in,
                              void* d_out, int out_size, void* d_ws, size_t ws_size,
                              hipStream_t stream) {
  const float* x   = (const float*)d_in[0];
  const int*   dis = (const int*)  d_in[1];
  const float* Ws  = (const float*)d_in[2];
  const float* bs  = (const float*)d_in[3];
  const float* Wp  = (const float*)d_in[4];
  const float* bp  = (const float*)d_in[5];
  const float* Wt  = (const float*)d_in[6];
  const float* bt  = (const float*)d_in[7];
  const float* Wg  = (const float*)d_in[8];
  const float* bg  = (const float*)d_in[9];
  const float* Wh  = (const float*)d_in[10];
  const float* bh  = (const float*)d_in[11];
  float* out = (float*)d_out;

  // workspace layout
  char* w = (char*)d_ws;
  size_t off = 0;
  unsigned short* F = (unsigned short*)(w + off); off += (size_t)NROWS * 3072 * 2;          // 25.2MB
  float* ht     = (float*)(w + off); off += (size_t)NROWS * TOPK * DOUT * 4;                // 33.6MB
  float* logits = (float*)(w + off); off += (size_t)NROWS * NTR * 4;
  float* wtop   = (float*)(w + off); off += (size_t)NROWS * TOPK * 4;
  int*   idxtop = (int*)  (w + off); off += (size_t)NROWS * TOPK * 4;
  int*   listT  = (int*)  (w + off); off += (size_t)LIST_T_CAP * 4;
  float* wlistT = (float*)(w + off); off += (size_t)LIST_T_CAP * 4;
  int*   listP  = (int*)  (w + off); off += (size_t)LIST_P_CAP * 4;
  int*   meta   = (int*)  (w + off); off += (size_t)META_N * 4;
  (void)ws_size; (void)in_sizes; (void)n_in; (void)out_size;

  // 1) routing
  init_routing<<<64, 256, 0, stream>>>(listT, listP, meta);
  gate_logits<<<(NROWS * NTR) / 256, 256, 0, stream>>>(x, Wg, bg, logits);
  route_kernel<<<NROWS / 256, 256, 0, stream>>>(logits, dis, wtop, idxtop, meta);
  offsets_kernel<<<1, 32, 0, stream>>>(meta);
  scatter_kernel<<<NROWS / 256, 256, 0, stream>>>(wtop, idxtop, dis, listT, wlistT, listP, meta);

  // 2) shared branch -> F[:, 0:1024]
  gemm64<0><<<dim3(NROWS / 64, DOUT / 64), 256, 0, stream>>>(
      x, Ws, bs, F, nullptr, nullptr, nullptr, 3072);

  // 3) private branch (gathered by disease) -> F[:, 1024:2048]
  gemm64<1><<<dim3(LIST_P_CAP / 64, DOUT / 64), 256, 0, stream>>>(
      x, Wp, bp, F + 1024, listP, nullptr, meta + OFF_P, 3072);

  // 4) transfer branch (gathered by expert, per-slot weighted) -> ht
  gemm64<2><<<dim3(LIST_T_CAP / 64, DOUT / 64), 256, 0, stream>>>(
      x, Wt, bt, ht, listT, wlistT, meta + OFF_T, 1024);

  // 5) combine slots -> F[:, 2048:3072]
  combine_kernel<<<(NROWS * DOUT) / 256, 256, 0, stream>>>(ht, F);

  // 6) heads: u and logvar, K=3072 over stacked features / stacked W_heads
  gemm64<3><<<dim3(NROWS / 64, DOUT / 64), 256, 0, stream>>>(
      F, Wh, bh, out, nullptr, nullptr, nullptr, 1024);
  gemm64<3><<<dim3(NROWS / 64, DOUT / 64), 256, 0, stream>>>(
      F, Wh + 3ull * 1024 * 1024, bh + 3 * 1024, out + (size_t)NROWS * DOUT,
      nullptr, nullptr, nullptr, 1024);
}
